// Mamba_77816217469377
// MI455X (gfx1250) — matrix operand order repaired and checked by reading, not compiled, not run on hardware
//
#include <hip/hip_runtime.h>
#include <hip/hip_bf16.h>
#include <math.h>

typedef __attribute__((ext_vector_type(16))) _Float16 v16h;
typedef __attribute__((ext_vector_type(8)))  _Float16 v8h;
typedef __attribute__((ext_vector_type(4)))  _Float16 v4h;
typedef __attribute__((ext_vector_type(8)))  float    v8f;

union V16U { v16h v; v8h h[2]; };

static constexpr int D_MODEL  = 512;
static constexpr int D_STATE  = 16;
static constexpr int D_INNER  = 1024;
static constexpr int N_HEADS  = 8;
static constexpr int HEAD_DIM = 128;
static constexpr int DT_RANK  = 32;
static constexpr int OUT_DIM  = 560;   // 32 + 4*128 + 16
static constexpr int BSZ      = 2;
static constexpr int LSEQ     = 1024;
static constexpr int MROWS    = BSZ * LSEQ;   // 2048

__device__ __forceinline__ float sigmoidf_(float x) { return 1.f / (1.f + __expf(-x)); }
__device__ __forceinline__ float siluf_(float x)    { return x * sigmoidf_(x); }
__device__ __forceinline__ float softplusf_(float x){ return (x > 20.f) ? x : log1pf(__expf(x)); }

// ---------------------------------------------------------------------------
// WMMA GEMM:  C[M,N] = A[M,K] (row major) * W[N,K]^T (row major weights)
// Block = 256 threads = 8 waves; block tile 64(M) x 32(N); K stepped by 32.
// Double-buffered LDS: one barrier per K-step; next tile's global loads
// overlap the current v_wmma_f32_16x16x32_f16.
// All staging is branch-free (clamped row + multiply mask) and vectorized:
//   A: 2x global_load_b128 -> v8h -> ds_store_b128   (8 elems/thread)
//   B: 1x global_load_b128 -> v4h -> ds_store_b64    (4 elems/thread)
// Fragment lane maps per the CDNA5 ISA 16-bit A(16x32)/B(32x16) layouts.
// AMODE==1: A element = A[m,k] * silu(A2[m,k])   (fused gating for out_proj)
// EPI==1:   C = softplus(acc + bias[n])          (fused dt_proj epilogue)
// ---------------------------------------------------------------------------
template <int AMODE, int EPI>
__global__ __launch_bounds__(256) void wmma_gemm_kernel(
    const float* __restrict__ A, const float* __restrict__ A2,
    const float* __restrict__ W, const float* __restrict__ bias,
    float* __restrict__ Cout,
    int M, int N, int K, int lda, int lda2, int ldc)
{
    __shared__ __align__(16) _Float16 Ah[2][64][40];   // 64 rows x 32 K (+pad)
    __shared__ __align__(16) _Float16 Bh[2][32][40];   // 32 n   x 32 K (+pad)

    const int t      = threadIdx.x;
    const int m_base = blockIdx.x * 64;
    const int n_base = blockIdx.y * 32;

    const int wave = t >> 5, lane = t & 31;
    const int m_tile = wave & 3;        // 4 M-tiles
    const int n_tile = wave >> 2;       // 2 N-tiles
    const int hs = lane >> 4;           // half-wave select
    const int r  = lane & 15;

    // staging indices (uniform per thread)
    const int sa_row = t >> 2;              // A: 4 threads per row
    const int sa_kq  = (t & 3) * 8;
    const int sb_nl  = t >> 3;              // B: 8 threads per n-row
    const int sb_kq  = (t & 7) * 4;
    const int sb_n   = n_base + sb_nl;
    const int sb_nc  = (sb_n < N) ? sb_n : (N - 1);      // clamp: always valid
    const float sb_msk = (sb_n < N) ? 1.f : 0.f;

    auto stage = [&](int buf, int k0) {
        // ---- A tile ----
        {
            const float4* ap =
                (const float4*)(A + (size_t)(m_base + sa_row) * lda + k0 + sa_kq);
            float4 a0 = ap[0], a1 = ap[1];
            if (AMODE == 1) {
                const float4* zp =
                    (const float4*)(A2 + (size_t)(m_base + sa_row) * lda2 + k0 + sa_kq);
                const float4 z0 = zp[0], z1 = zp[1];
                a0.x *= siluf_(z0.x); a0.y *= siluf_(z0.y);
                a0.z *= siluf_(z0.z); a0.w *= siluf_(z0.w);
                a1.x *= siluf_(z1.x); a1.y *= siluf_(z1.y);
                a1.z *= siluf_(z1.z); a1.w *= siluf_(z1.w);
            }
            v8h hv;
            hv[0] = (_Float16)a0.x; hv[1] = (_Float16)a0.y;
            hv[2] = (_Float16)a0.z; hv[3] = (_Float16)a0.w;
            hv[4] = (_Float16)a1.x; hv[5] = (_Float16)a1.y;
            hv[6] = (_Float16)a1.z; hv[7] = (_Float16)a1.w;
            *(v8h*)&Ah[buf][sa_row][sa_kq] = hv;
        }
        // ---- B tile (stored [n][k]) ----
        {
            const float4 wv = *(const float4*)(W + (size_t)sb_nc * K + k0 + sb_kq);
            v4h hw;
            hw[0] = (_Float16)(wv.x * sb_msk);
            hw[1] = (_Float16)(wv.y * sb_msk);
            hw[2] = (_Float16)(wv.z * sb_msk);
            hw[3] = (_Float16)(wv.w * sb_msk);
            *(v4h*)&Bh[buf][sb_nl][sb_kq] = hw;
        }
    };

    v8f acc = {};

    stage(0, 0);
    __syncthreads();

    int cur = 0;
    for (int k0 = 0; k0 < K; k0 += 32) {
        // fragment loads from current buffer
        V16U a, b;
        const int arow = m_tile * 16 + r;
        a.h[0] = *(const v8h*)&Ah[cur][arow][hs * 8];        // K 0..7  / 8..15
        a.h[1] = *(const v8h*)&Ah[cur][arow][16 + hs * 8];   // K 16..23/ 24..31
        const int bcol = n_tile * 16 + r;
        b.h[0] = *(const v8h*)&Bh[cur][bcol][hs * 8];        // K 0..7  / 8..15
        b.h[1] = *(const v8h*)&Bh[cur][bcol][16 + hs * 8];   // K 16..23/ 24..31

        // prefetch next tile into the other buffer (uniform branch)
        if (k0 + 32 < K) stage(cur ^ 1, k0 + 32);

        acc = __builtin_amdgcn_wmma_f32_16x16x32_f16(
            /*neg_a=*/false, a.v, /*neg_b=*/false, b.v,
            /*c_mod=*/(short)0, acc, /*reuse_a=*/false, /*reuse_b=*/false);

        __syncthreads();
        cur ^= 1;
    }

    // C/D layout: VGPR i -> M = m_tile*16 + hs*8 + i, N = n_tile*16 + r
    const int n = n_base + n_tile * 16 + r;
    if (n < N) {
        #pragma unroll
        for (int i = 0; i < 8; ++i) {
            const int m = m_base + m_tile * 16 + hs * 8 + i;
            float v = acc[i];
            if (EPI == 1) v = softplusf_(v + bias[n]);
            Cout[(size_t)m * ldc + n] = v;
        }
    }
}

// ---------------------------------------------------------------------------
// Causal depthwise conv (k=3, left pad 2) + bias + SiLU over xp = xz[:, :1024]
// ---------------------------------------------------------------------------
__global__ __launch_bounds__(256) void conv_silu_kernel(
    const float* __restrict__ xz, const float* __restrict__ conv_w,
    const float* __restrict__ conv_b, float* __restrict__ xc)
{
    const int idx = blockIdx.x * 256 + threadIdx.x;      // MROWS * D_INNER
    const int c = idx & (D_INNER - 1);
    const int m = idx >> 10;
    const int l = m & (LSEQ - 1);
    const float w0 = conv_w[c * 3 + 0], w1 = conv_w[c * 3 + 1], w2 = conv_w[c * 3 + 2];
    const float x0 = (l >= 2) ? xz[(size_t)(m - 2) * (2 * D_INNER) + c] : 0.f;
    const float x1 = (l >= 1) ? xz[(size_t)(m - 1) * (2 * D_INNER) + c] : 0.f;
    const float x2 = xz[(size_t)m * (2 * D_INNER) + c];
    xc[idx] = siluf_(w0 * x0 + w1 * x1 + w2 * x2 + conv_b[c]);
}

// ---------------------------------------------------------------------------
// Complex SSM scan. Grid (d_chunk=8, h=8, b=2); 256 threads = 16 d x 16 n.
// State h[d,n] is register-resident; sequential over L; n-reduction via
// __shfl_xor butterflies confined to 16-lane halves of the wave32.
// ---------------------------------------------------------------------------
__global__ __launch_bounds__(256) void ssm_scan_kernel(
    const float* __restrict__ proj, const float* __restrict__ delta,
    const float* __restrict__ xc, const float* __restrict__ A_log,
    const float* __restrict__ A_imag, const float* __restrict__ Dvec,
    float* __restrict__ y)
{
    const int dchunk = blockIdx.x;
    const int h      = blockIdx.y;
    const int b      = blockIdx.z;
    const int t  = threadIdx.x;
    const int dl = t >> 4;
    const int n  = t & 15;
    const int d  = dchunk * 16 + dl;
    const int c  = h * HEAD_DIM + d;

    const float a_re = -__expf(A_log[h * D_STATE + n]);
    const float a_im = A_imag[h * D_STATE + n];
    const float Dv   = Dvec[c];

    float h_re = 0.f, h_im = 0.f;
    float bxp_re = 0.f, bxp_im = 0.f;   // Bx at t-1 (zero-padded at t=0)

    for (int l = 0; l < LSEQ; ++l) {
        const size_t m  = (size_t)b * LSEQ + l;
        const float* pr = proj + m * OUT_DIM;
        const float dt_v = delta[m * D_INNER + c];
        const float x_v  = xc[m * D_INNER + c];
        const float b_re = pr[ 32 + h * 16 + n];
        const float b_im = pr[160 + h * 16 + n];
        const float c_re = pr[288 + h * 16 + n];
        const float c_im = pr[416 + h * 16 + n];
        const float lg = sigmoidf_(pr[544 + h]);
        const float eg = sigmoidf_(pr[552 + h]);

        // alpha = exp(clip(dt*A.re) + i*dt*A.im)
        const float dre = fminf(fmaxf(dt_v * a_re, -20.f), 20.f);
        const float dim = dt_v * a_im;
        const float ea  = __expf(dre);
        float s, cc;
        __sincosf(dim, &s, &cc);
        const float al_re = ea * cc, al_im = ea * s;

        const float bx_re = x_v * b_re, bx_im = x_v * b_im;
        const float bs    = (1.f - lg) * dt_v;          // beta = bs * alpha
        const float be_re = bs * al_re, be_im = bs * al_im;
        const float g     = lg * dt_v * eg;             // gamma (real)

        const float u_re = be_re * bxp_re - be_im * bxp_im + g * bx_re;
        const float u_im = be_re * bxp_im + be_im * bxp_re + g * bx_im;

        const float nh_re = al_re * h_re - al_im * h_im + u_re;
        const float nh_im = al_re * h_im + al_im * h_re + u_im;
        h_re = nh_re; h_im = nh_im;
        bxp_re = bx_re; bxp_im = bx_im;

        float yv = h_re * c_re + h_im * c_im;           // Re(h * conj(C))
        yv += __shfl_xor(yv, 1);
        yv += __shfl_xor(yv, 2);
        yv += __shfl_xor(yv, 4);
        yv += __shfl_xor(yv, 8);
        if (n == 0) y[m * D_INNER + c] = yv + Dv * x_v;
    }
}

// ---------------------------------------------------------------------------
extern "C" void kernel_launch(void* const* d_in, const int* in_sizes, int n_in,
                              void* d_out, int out_size, void* d_ws, size_t ws_size,
                              hipStream_t stream)
{
    const float* x        = (const float*)d_in[0];
    const float* in_proj  = (const float*)d_in[1];   // (2048, 512)
    const float* conv_w   = (const float*)d_in[2];   // (1024, 1, 3)
    const float* conv_b   = (const float*)d_in[3];
    const float* x_proj_w = (const float*)d_in[4];   // (560, 1024)
    const float* dt_w     = (const float*)d_in[5];   // (1024, 32)
    const float* dt_b     = (const float*)d_in[6];
    const float* A_log    = (const float*)d_in[7];
    const float* A_imag   = (const float*)d_in[8];
    const float* Dvec     = (const float*)d_in[9];
    const float* out_w    = (const float*)d_in[10];  // (512, 1024)
    float* out = (float*)d_out;

    float* ws    = (float*)d_ws;
    float* xz    = ws;                                   // 2048 x 2048
    float* xc    = xz    + (size_t)MROWS * 2 * D_INNER;  // 2048 x 1024
    float* proj  = xc    + (size_t)MROWS * D_INNER;      // 2048 x 560
    float* delta = proj  + (size_t)MROWS * OUT_DIM;      // 2048 x 1024
    float* y     = delta + (size_t)MROWS * D_INNER;      // 2048 x 1024

    const dim3 blk(256);

    // 1) xz = x @ in_proj_w.T        (M=2048, N=2048, K=512)
    wmma_gemm_kernel<0, 0><<<dim3(MROWS / 64, (2 * D_INNER) / 32), blk, 0, stream>>>(
        x, nullptr, in_proj, nullptr, xz,
        MROWS, 2 * D_INNER, D_MODEL, D_MODEL, 0, 2 * D_INNER);

    // 2) xc = silu(depthwise_conv(xp) + b)
    conv_silu_kernel<<<(MROWS * D_INNER) / 256, blk, 0, stream>>>(xz, conv_w, conv_b, xc);

    // 3) proj = xc @ x_proj_w.T      (M=2048, N=560, K=1024)
    wmma_gemm_kernel<0, 0><<<dim3(MROWS / 64, (OUT_DIM + 31) / 32), blk, 0, stream>>>(
        xc, nullptr, x_proj_w, nullptr, proj,
        MROWS, OUT_DIM, D_INNER, D_INNER, 0, OUT_DIM);

    // 4) delta = softplus(delta_raw @ dt_proj_w.T + dt_b)  (M=2048, N=1024, K=32)
    //    delta_raw = first 32 columns of proj (lda = 560)
    wmma_gemm_kernel<0, 1><<<dim3(MROWS / 64, D_INNER / 32), blk, 0, stream>>>(
        proj, nullptr, dt_w, dt_b, delta,
        MROWS, D_INNER, DT_RANK, OUT_DIM, 0, D_INNER);

    // 5) complex SSM scan -> y (includes + D * xc)
    ssm_scan_kernel<<<dim3(HEAD_DIM / 16, N_HEADS, BSZ), blk, 0, stream>>>(
        proj, delta, xc, A_log, A_imag, Dvec, y);

    // 6) out = (y * silu(z)) @ out_proj_w.T   (M=2048, N=512, K=1024), z = xz[:,1024:]
    wmma_gemm_kernel<1, 0><<<dim3(MROWS / 64, D_MODEL / 32), blk, 0, stream>>>(
        y, xz + D_INNER, out_w, nullptr, out,
        MROWS, D_MODEL, D_INNER, D_INNER, 2 * D_INNER, D_MODEL);
}
